// GrayInspectH_52201032516082
// MI455X (gfx1250) — matrix operand order repaired and checked by reading, not compiled, not run on hardware
//
#include <hip/hip_runtime.h>
#include <hip/hip_bf16.h>

// ---------------- types ----------------
typedef __bf16 bf16_t;
typedef float v8f  __attribute__((ext_vector_type(8)));
typedef bf16_t v16bf __attribute__((ext_vector_type(16)));

union FragU {
  v16bf v;
  uint4 u[2];
};

__device__ __forceinline__ v8f wmma_bf16(v16bf a, v16bf b, v8f c) {
  // D = A(16x32 bf16) * B(32x16 bf16) + C(16x16 f32)
  return __builtin_amdgcn_wmma_f32_16x16x32_bf16(
      false, a, false, b, (short)0, c, false, false);
}

__device__ __forceinline__ unsigned pk2(float a, float b) {
  unsigned short ua = __builtin_bit_cast(unsigned short, (bf16_t)a);
  unsigned short ub = __builtin_bit_cast(unsigned short, (bf16_t)b);
  return (unsigned)ua | ((unsigned)ub << 16);
}

// gfx1250 async global->LDS copy (ASYNCcnt path). LDS offset = low 32 bits of
// the flat pointer (generic LDS pointers are {aperture_hi, lds_offset}).
__device__ __forceinline__ void async_b128(void* lds, const void* g) {
  unsigned loff = (unsigned)(unsigned long long)(uintptr_t)lds;
  asm volatile("global_load_async_to_lds_b128 %0, %1, off"
               :: "v"(loff), "v"((unsigned long long)(uintptr_t)g)
               : "memory");
}

// ---------------- problem constants ----------------
#define BATCH   8
#define NHEADS  8
#define SEQ     1024
#define HD      64
#define DIM     512
#define NQKV    1536
#define MTOT    (BATCH * SEQ)        // 8192
#define MAXHW   32

// ---------------- prep kernels ----------------
__global__ __launch_bounds__(256) void cvt_f32_bf16(const float* __restrict__ in,
                                                    bf16_t* __restrict__ out, int n) {
  int i = blockIdx.x * 256 + threadIdx.x;
  if (i < n) out[i] = (bf16_t)in[i];
}

// W[K,N] (row-major) -> WT[N,K] bf16
__global__ __launch_bounds__(256) void transpose_cvt(const float* __restrict__ W,
                                                     bf16_t* __restrict__ WT,
                                                     int K, int N, int total) {
  int i = blockIdx.x * 256 + threadIdx.x;
  if (i >= total) return;
  int n = i / K;
  int k = i - n * K;
  WT[i] = (bf16_t)W[(size_t)k * N + n];
}

// token_mask [B,1024] (bool bytes) -> bit words [B,32]
__global__ __launch_bounds__(256) void pack_mask(const unsigned char* __restrict__ m,
                                                 unsigned* __restrict__ out) {
  int i = threadIdx.x;  // 256 = B*32 words
  int b = i >> 5, wd = i & 31;
  unsigned v = 0;
#pragma unroll
  for (int j = 0; j < 32; ++j)
    v |= (m[b * SEQ + wd * 32 + j] ? 1u : 0u) << j;
  out[i] = v;
}

// ---------------- tiled bf16 GEMM with WMMA + async double-buffered LDS ----
// C[M,N] = A[M,K] * BT[N,K]^T + bias[N]
// EPI 0: scatter into Q/K [B,nh,L,64] and Vt [B,nh,64,L] (bf16)
// EPI 1: write f32 to outF[M,N]
template <int EPI>
__global__ __launch_bounds__(256) void gemm_bf16(const bf16_t* __restrict__ A,
                                                 const bf16_t* __restrict__ BT,
                                                 const float* __restrict__ bias,
                                                 float* __restrict__ outF,
                                                 bf16_t* __restrict__ Qo,
                                                 bf16_t* __restrict__ Ko,
                                                 bf16_t* __restrict__ Vto,
                                                 int N, int K) {
  __shared__ __attribute__((aligned(32))) bf16_t As[2][128 * 32];
  __shared__ __attribute__((aligned(32))) bf16_t Bs[2][128 * 32];

  const int tid = threadIdx.x;
  const int w   = tid >> 5;
  const int l   = tid & 31;
  const int g   = l >> 4;          // half-group (0/1)
  const int ln  = l & 15;          // lane within half-group
  const int mBase = blockIdx.y * 128;
  const int nBase = blockIdx.x * 128;
  const int wm = (w >> 2) * 64;    // wave M offset (2 waves in M)
  const int wn = (w & 3) * 32;     // wave N offset (4 waves in N)

  const v8f zero8 = {0.f, 0.f, 0.f, 0.f, 0.f, 0.f, 0.f, 0.f};
  v8f acc[4][2];
#pragma unroll
  for (int mt = 0; mt < 4; ++mt)
#pragma unroll
    for (int nt = 0; nt < 2; ++nt) acc[mt][nt] = zero8;

  const int fillRow = tid >> 1;
  const int fillCol = (tid & 1) * 16;
  const bf16_t* Arow = A + (size_t)(mBase + fillRow) * K + fillCol;
  const bf16_t* Brow = BT + (size_t)(nBase + fillRow) * K + fillCol;
  bf16_t* asDst0 = &As[0][fillRow * 32 + fillCol];
  bf16_t* bsDst0 = &Bs[0][fillRow * 32 + fillCol];
  bf16_t* asDst1 = &As[1][fillRow * 32 + fillCol];
  bf16_t* bsDst1 = &Bs[1][fillRow * 32 + fillCol];

  // prologue: stage tile 0
  async_b128(asDst0, Arow);
  async_b128(asDst0 + 8, Arow + 8);
  async_b128(bsDst0, Brow);
  async_b128(bsDst0 + 8, Brow + 8);

  const int KT = K >> 5;
  for (int kt = 0; kt < KT; ++kt) {
    const int cur = kt & 1;
    // issue next tile into the other buffer (4 async ops)
    if (kt + 1 < KT) {
      const int k0n = (kt + 1) * 32;
      bf16_t* ad = cur ? asDst0 : asDst1;
      bf16_t* bd = cur ? bsDst0 : bsDst1;
      async_b128(ad, Arow + k0n);
      async_b128(ad + 8, Arow + k0n + 8);
      async_b128(bd, Brow + k0n);
      async_b128(bd + 8, Brow + k0n + 8);
      asm volatile("s_wait_asynccnt 0x4" ::: "memory");  // current tile done
    } else {
      asm volatile("s_wait_asynccnt 0x0" ::: "memory");
    }
    __syncthreads();

    const bf16_t* Ab = As[cur];
    const bf16_t* Bb = Bs[cur];
    v16bf af[4], bfr[2];
#pragma unroll
    for (int mt = 0; mt < 4; ++mt) {
      const int row = wm + mt * 16 + ln;
      FragU fu;
      fu.u[0] = *reinterpret_cast<const uint4*>(&Ab[row * 32 + 8 * g]);
      fu.u[1] = *reinterpret_cast<const uint4*>(&Ab[row * 32 + 16 + 8 * g]);
      af[mt] = fu.v;
    }
#pragma unroll
    for (int nt = 0; nt < 2; ++nt) {
      const int row = wn + nt * 16 + ln;
      FragU fu;
      fu.u[0] = *reinterpret_cast<const uint4*>(&Bb[row * 32 + 8 * g]);
      fu.u[1] = *reinterpret_cast<const uint4*>(&Bb[row * 32 + 16 + 8 * g]);
      bfr[nt] = fu.v;
    }
    __syncthreads();  // readers done before next overwrite of this buffer

#pragma unroll
    for (int mt = 0; mt < 4; ++mt)
#pragma unroll
      for (int nt = 0; nt < 2; ++nt)
        acc[mt][nt] = wmma_bf16(af[mt], bfr[nt], acc[mt][nt]);
  }

  // epilogue — C/D layout: lane col n = ln, row = i + 8g
#pragma unroll
  for (int mt = 0; mt < 4; ++mt) {
#pragma unroll
    for (int nt = 0; nt < 2; ++nt) {
      const int col = nBase + wn + nt * 16 + ln;
      const float bv = bias[col];
#pragma unroll
      for (int i = 0; i < 8; ++i) {
        const int row = mBase + wm + mt * 16 + i + 8 * g;
        const float val = acc[mt][nt][i] + bv;
        if (EPI == 0) {
          const int which = col >> 9;          // 0:q 1:k 2:v
          const int h = (col >> 6) & 7;
          const int d = col & 63;
          const int b = row >> 10;
          const int li = row & 1023;
          const size_t bh = (size_t)(b * NHEADS + h);
          if (which == 0)
            Qo[(bh * SEQ + li) * HD + d] = (bf16_t)val;
          else if (which == 1)
            Ko[(bh * SEQ + li) * HD + d] = (bf16_t)val;
          else
            Vto[(bh * HD + d) * SEQ + li] = (bf16_t)val;   // V transposed
        } else {
          outF[(size_t)row * N + col] = val;
        }
      }
    }
  }
}

// ---------------- flash attention kernel ----------------
// grid: B*nh*16 blocks, 128 threads (4 waves); wave handles 16 query rows.
// Softmax is done in a transposed domain (one row per lane) to avoid long
// ds_bpermute chains: only 2 cross-lane shuffles per 32-key block.
__global__ __launch_bounds__(128) void attn_kernel(const bf16_t* __restrict__ Q,
                                                   const bf16_t* __restrict__ Kb,
                                                   const bf16_t* __restrict__ Vt,
                                                   const float* __restrict__ hbias,
                                                   const float* __restrict__ wbias,
                                                   const unsigned* __restrict__ Mbits,
                                                   bf16_t* __restrict__ Ctx) {
  __shared__ __attribute__((aligned(16))) float  Sst[4 * 512];   // raw scores
  __shared__ __attribute__((aligned(32))) bf16_t Pst[4 * 512];   // exp'ed P
  __shared__ __attribute__((aligned(16))) float  bwT[4 * 512];   // width-bias LUT
  __shared__ __attribute__((aligned(16))) float  alphaA[64];
  __shared__ __attribute__((aligned(16))) float  linvA[64];

  const int tid = threadIdx.x;
  const int w = tid >> 5, l = tid & 31;
  const int g = l >> 4, ln = l & 15;

  const int blk  = blockIdx.x;
  const int bhid = blk >> 4;           // b*8+h
  const int qblk = blk & 15;
  const int b = bhid >> 3, h = bhid & 7;
  const int qt = qblk * 4 + w;         // this wave's 16-query tile

  // per-wave width-bias table: bwT[r][kw] = wbias[(qw(r)-kw+31)*8+h]
#pragma unroll
  for (int t = 0; t < 16; ++t) {
    const int idx = t * 32 + l;
    const int rr = idx >> 5, kw = idx & 31;
    const int qw = (qt * 16 + rr) & 31;
    bwT[w * 512 + idx] = wbias[(qw - kw + 31) * NHEADS + h];
  }

  const bf16_t* Qp = Q  + (size_t)bhid * SEQ * HD;
  const bf16_t* Kp = Kb + (size_t)bhid * SEQ * HD;
  const bf16_t* Vp = Vt + (size_t)bhid * HD * SEQ;
  const unsigned* mrow = Mbits + b * 32;

  // Q A-fragments (16x32, two K-steps over hd=64); loaded once
  v16bf qf[2];
#pragma unroll
  for (int s = 0; s < 2; ++s) {
    const int qrow = qt * 16 + ln;
    FragU fu;
    fu.u[0] = *reinterpret_cast<const uint4*>(Qp + (size_t)qrow * HD + 32 * s + 8 * g);
    fu.u[1] = *reinterpret_cast<const uint4*>(Qp + (size_t)qrow * HD + 32 * s + 8 * g + 16);
    qf[s] = fu.v;
  }

  const v8f zero8 = {0.f, 0.f, 0.f, 0.f, 0.f, 0.f, 0.f, 0.f};
  v8f o[4];
#pragma unroll
  for (int j = 0; j < 4; ++j) o[j] = zero8;

  // transposed-domain softmax state: this lane owns row (l&15); the two
  // half-groups hold identical copies.
  float m_p = -1.0e30f, l_p = 0.f;
  const int qrow_t = qt * 16 + ln;
  const int qh_t = qrow_t >> 5;
  const float scale = 0.125f;  // hd^-0.5

  for (int kb = 0; kb < SEQ / 32; ++kb) {
    const int k0 = kb * 32;
    v8f s0 = zero8, s1 = zero8;

    // S(16x32) = Q * K^T : 4 WMMAs
#pragma unroll
    for (int s = 0; s < 2; ++s) {
      FragU f0, f1;
      {
        const int key = k0 + ln;
        f0.u[0] = *reinterpret_cast<const uint4*>(Kp + (size_t)key * HD + 32 * s + 8 * g);
        f0.u[1] = *reinterpret_cast<const uint4*>(Kp + (size_t)key * HD + 32 * s + 8 * g + 16);
      }
      {
        const int key = k0 + 16 + ln;
        f1.u[0] = *reinterpret_cast<const uint4*>(Kp + (size_t)key * HD + 32 * s + 8 * g);
        f1.u[1] = *reinterpret_cast<const uint4*>(Kp + (size_t)key * HD + 32 * s + 8 * g + 16);
      }
      s0 = wmma_bf16(qf[s], f0.v, s0);
      s1 = wmma_bf16(qf[s], f1.v, s1);
    }

    // stage raw S (C-layout lanes -> [row][col] tile)
#pragma unroll
    for (int i = 0; i < 8; ++i) {
      const int row = i + 8 * g;
      Sst[w * 512 + row * 32 + ln]      = s0[i];
      Sst[w * 512 + row * 32 + 16 + ln] = s1[i];
    }
    asm volatile("s_wait_dscnt 0x0" ::: "memory");

    // ---- transposed domain: lane owns row ln, columns g*16..g*16+15 ----
    const float* Srow = &Sst[w * 512 + ln * 32 + g * 16];
    const float* Brow = &bwT[w * 512 + ln * 32 + g * 16];
    const float bh = hbias[(qh_t - kb + 31) * NHEADS + h];  // kh == kb here
    const unsigned mw = mrow[kb];

    float sv[16];
#pragma unroll
    for (int c = 0; c < 16; ++c) {
      const float x = Srow[c] * scale + bh + Brow[c];
      const bool mk = (mw >> (g * 16 + c)) & 1u;
      sv[c] = mk ? x : -10000.f;
    }
    float mx = sv[0];
#pragma unroll
    for (int c = 1; c < 16; ++c) mx = fmaxf(mx, sv[c]);
    mx = fmaxf(mx, __shfl_xor(mx, 16, 32));          // combine the two halves
    const float mnew = fmaxf(m_p, mx);
    const float al = __expf(m_p - mnew);
    m_p = mnew;
    float ss = 0.f;
#pragma unroll
    for (int c = 0; c < 16; ++c) {
      sv[c] = __expf(sv[c] - mnew);
      ss += sv[c];
    }
    ss += __shfl_xor(ss, 16, 32);
    l_p = l_p * al + ss;
    alphaA[w * 16 + ln] = al;                         // both halves: same value

    // pack P row (bf16) directly in A-fragment-friendly [row][col] layout
    uint4 p0, p1;
    p0.x = pk2(sv[0], sv[1]);  p0.y = pk2(sv[2], sv[3]);
    p0.z = pk2(sv[4], sv[5]);  p0.w = pk2(sv[6], sv[7]);
    p1.x = pk2(sv[8], sv[9]);  p1.y = pk2(sv[10], sv[11]);
    p1.z = pk2(sv[12], sv[13]); p1.w = pk2(sv[14], sv[15]);
    uint4* pdst = reinterpret_cast<uint4*>(&Pst[w * 512 + ln * 32 + g * 16]);
    pdst[0] = p0;
    pdst[1] = p1;
    asm volatile("s_wait_dscnt 0x0" ::: "memory");

    // ---- back to C-layout: rescale O, then P@V ----
    float av[8];
#pragma unroll
    for (int i = 0; i < 8; ++i) av[i] = alphaA[w * 16 + 8 * g + i];
#pragma unroll
    for (int j = 0; j < 4; ++j)
#pragma unroll
      for (int i = 0; i < 8; ++i) o[j][i] *= av[i];

    FragU pf;
    pf.u[0] = *reinterpret_cast<const uint4*>(&Pst[w * 512 + ln * 32 + 8 * g]);
    pf.u[1] = *reinterpret_cast<const uint4*>(&Pst[w * 512 + ln * 32 + 16 + 8 * g]);

#pragma unroll
    for (int j = 0; j < 4; ++j) {
      const int d = j * 16 + ln;
      FragU vf;
      vf.u[0] = *reinterpret_cast<const uint4*>(Vp + (size_t)d * SEQ + k0 + 8 * g);
      vf.u[1] = *reinterpret_cast<const uint4*>(Vp + (size_t)d * SEQ + k0 + 8 * g + 16);
      o[j] = wmma_bf16(pf.v, vf.v, o[j]);
    }
  }

  // finalize: 1/l with query-mask folded, broadcast via LDS to C-layout lanes
  {
    const unsigned qw = mrow[qrow_t >> 5];
    const float qm = ((qw >> (qrow_t & 31)) & 1u) ? 1.f : 0.f;
    linvA[w * 16 + ln] = (l_p > 0.f) ? (qm / l_p) : 0.f;
  }
  asm volatile("s_wait_dscnt 0x0" ::: "memory");
#pragma unroll
  for (int i = 0; i < 8; ++i) {
    const int qrow = qt * 16 + i + 8 * g;
    const float inv = linvA[w * 16 + 8 * g + i];
    bf16_t* dst = Ctx + ((size_t)(b * SEQ + qrow)) * DIM + h * HD;
#pragma unroll
    for (int j = 0; j < 4; ++j) dst[j * 16 + ln] = (bf16_t)(o[j][i] * inv);
  }
}

// ---------------- launcher ----------------
extern "C" void kernel_launch(void* const* d_in, const int* in_sizes, int n_in,
                              void* d_out, int out_size, void* d_ws, size_t ws_size,
                              hipStream_t stream) {
  const float*         tokens = (const float*)d_in[0];
  const unsigned char* tmask  = (const unsigned char*)d_in[1];
  const float*         Wqkv   = (const float*)d_in[2];
  const float*         bqkv   = (const float*)d_in[3];
  const float*         Wout   = (const float*)d_in[4];
  const float*         bout   = (const float*)d_in[5];
  const float*         hb     = (const float*)d_in[6];
  const float*         wb     = (const float*)d_in[7];
  float* out = (float*)d_out;

  char* ws = (char*)d_ws;
  bf16_t*   X      = (bf16_t*)(ws + 0);           //  8 MB  tokens bf16 [8192,512]
  bf16_t*   WqkvT  = (bf16_t*)(ws + 8388608);     //  1.5MB [1536,512]
  bf16_t*   WoutT  = (bf16_t*)(ws + 9961472);     //  0.5MB [512,512]
  bf16_t*   Qb     = (bf16_t*)(ws + 10485760);    //  8 MB  [B,nh,L,64]
  bf16_t*   Kbuf   = (bf16_t*)(ws + 18874368);    //  8 MB  [B,nh,L,64]
  bf16_t*   Vt     = (bf16_t*)(ws + 27262976);    //  8 MB  [B,nh,64,L]
  bf16_t*   Ctx    = (bf16_t*)(ws + 35651584);    //  8 MB  [8192,512]
  unsigned* Mbits  = (unsigned*)(ws + 44040192);  //  1 KB  [B,32]

  // 1) precision conversion / weight transposes / mask packing
  cvt_f32_bf16<<<(MTOT * DIM + 255) / 256, 256, 0, stream>>>(tokens, X, MTOT * DIM);
  transpose_cvt<<<(DIM * NQKV + 255) / 256, 256, 0, stream>>>(Wqkv, WqkvT, DIM, NQKV, DIM * NQKV);
  transpose_cvt<<<(DIM * DIM + 255) / 256, 256, 0, stream>>>(Wout, WoutT, DIM, DIM, DIM * DIM);
  pack_mask<<<1, 256, 0, stream>>>(tmask, Mbits);

  // 2) QKV projection: [8192,512] x [512,1536] -> scattered Q/K/V
  gemm_bf16<0><<<dim3(NQKV / 128, MTOT / 128), 256, 0, stream>>>(
      X, WqkvT, bqkv, nullptr, Qb, Kbuf, Vt, NQKV, DIM);

  // 3) flash attention with relative position bias
  attn_kernel<<<BATCH * NHEADS * (SEQ / 64), 128, 0, stream>>>(
      Qb, Kbuf, Vt, hb, wb, Mbits, Ctx);

  // 4) output projection: [8192,512] x [512,512] + bout -> f32 out
  gemm_bf16<1><<<dim3(DIM / 128, MTOT / 128), 256, 0, stream>>>(
      Ctx, WoutT, bout, out, nullptr, nullptr, nullptr, DIM, DIM);
}
